// BatchGraphAttention_28003186770284
// MI455X (gfx1250) — matrix operand order repaired and checked by reading, not compiled, not run on hardware
//
#include <hip/hip_runtime.h>
#include <hip/hip_bf16.h>

// ---------------------------------------------------------------------------
// BatchGraphAttention for MI455X (gfx1250, wave32, WMMA).
// All GEMMs use v_wmma_f32_16x16x32_f16 (f16 inputs, f32 accumulate).
// adj (32MB, the HBM-dominant tensor) is read exactly once: one 16x32 adj
// tile per workgroup is shared across all 8 heads via LDS.
// Attention processes 32 keys/iteration so P@V WMMAs are fully dense.
// Global->LDS staging uses GLOBAL_LOAD_ASYNC_TO_LDS_B128 (ASYNCcnt path).
// ---------------------------------------------------------------------------

#define Bq 8
#define Nq 1024
#define Dq 512
#define Hq 8
#define HDq 64

typedef __attribute__((ext_vector_type(16))) _Float16 v16h;
typedef __attribute__((ext_vector_type(8)))  _Float16 v8h;
typedef __attribute__((ext_vector_type(8)))  float    v8f;
typedef int gv4i __attribute__((vector_size(16)));   // matches builtin's V4i

#if defined(__has_builtin)
#if __has_builtin(__builtin_amdgcn_global_load_async_to_lds_b128) && \
    __has_builtin(__builtin_amdgcn_s_wait_asynccnt)
#define USE_ASYNC_LDS 1
#endif
#endif

#define GPTR(p) ((__attribute__((address_space(1))) gv4i*)(void*)(size_t)(const void*)(p))
#define LPTR(p) ((__attribute__((address_space(3))) gv4i*)(void*)(p))

// ------------------------------ f32 -> f16 ---------------------------------
__global__ __launch_bounds__(256) void cvt_f32_f16(const float* __restrict__ in,
                                                   _Float16* __restrict__ out, int n) {
    int i = blockIdx.x * 256 + threadIdx.x;
    if (i < n) out[i] = (_Float16)in[i];
}

// ------------------------------ QKV projection -----------------------------
// y[m, n] = sum_k x16[m,k] * w16[n,k]   (Linear with W^T), y stored as f16.
// grid: (M/16, 512/128, 3) ; block 256 = 8 waves ; wave owns one 16x16 tile.
__global__ __launch_bounds__(256) void qkv_gemm(
    const _Float16* __restrict__ x16,
    const _Float16* __restrict__ wq16, const _Float16* __restrict__ wk16,
    const _Float16* __restrict__ wv16,
    _Float16* __restrict__ q16, _Float16* __restrict__ k16,
    _Float16* __restrict__ v16) {

    const _Float16* w   = (blockIdx.z == 0) ? wq16 : (blockIdx.z == 1) ? wk16 : wv16;
    _Float16*       out = (blockIdx.z == 0) ? q16  : (blockIdx.z == 1) ? k16  : v16;

    const int tid  = threadIdx.x;
    const int wave = tid >> 5;
    const int lane = tid & 31;
    const int hi   = lane >> 4;
    const int l16  = lane & 15;

    const int m0 = blockIdx.x * 16;                 // row tile in [0, B*N)
    const int n0 = blockIdx.y * 128 + wave * 16;    // col tile in [0, D)

    __shared__ __align__(16) _Float16 As[16 * Dq];  // 16KB A strip shared by 8 waves

    {   // cooperative stage of A strip (contiguous 16 rows x 512 halves)
        const char* src = (const char*)(x16 + (size_t)m0 * Dq);
#if USE_ASYNC_LDS
        for (int c = tid; c < 16 * Dq / 8; c += 256)
            __builtin_amdgcn_global_load_async_to_lds_b128(
                GPTR(src + 16 * c), LPTR((char*)As + 16 * c), 0, 0);
        __builtin_amdgcn_s_wait_asynccnt(0);
#else
        for (int c = tid; c < 16 * Dq / 8; c += 256)
            ((uint4*)As)[c] = ((const uint4*)src)[c];
#endif
    }
    __syncthreads();

    v8f acc = {};
    for (int kk = 0; kk < Dq; kk += 32) {
        // A fragment (contiguous k: element e of lane half hi holds k = 16*hi + e)
        const _Float16* ap = As + l16 * Dq + kk + hi * 16;
        v8h alo = *(const v8h*)ap;
        v8h ahi = *(const v8h*)(ap + 8);
        v16h a;
        #pragma unroll
        for (int e = 0; e < 8; ++e) { a[e] = alo[e]; a[e + 8] = ahi[e]; }
        // B fragment: lane holds a contiguous 16-half slice of column (n0+l16)
        v16h b = *(const v16h*)(w + (size_t)(n0 + l16) * Dq + kk + hi * 16);
        acc = __builtin_amdgcn_wmma_f32_16x16x32_f16(false, a, false, b,
                                                     (short)0, acc, false, false);
    }

    #pragma unroll
    for (int i = 0; i < 8; ++i) {
        int row = m0 + i + hi * 8;                  // C layout: VGPR i -> M = i + 8*hi
        out[(size_t)row * Dq + n0 + l16] = (_Float16)acc[i];
    }
}

// ------------------------------ attention ----------------------------------
// grid: (N/16 q-tiles, B) ; block 256 = 8 waves ; wave = head.
// Flash-style: stream 32 key-tiles of 32 keys, running max/sum, P@V via WMMA.
__global__ __launch_bounds__(256) void attn_kernel(
    const _Float16* __restrict__ q16, const _Float16* __restrict__ k16,
    const _Float16* __restrict__ v16, const int* __restrict__ adj,
    _Float16* __restrict__ attn16) {

    const int tid  = threadIdx.x;
    const int wave = tid >> 5;          // head
    const int lane = tid & 31;
    const int hi   = lane >> 4;
    const int l16  = lane & 15;
    const int h    = wave;
    const int b    = blockIdx.y;
    const int q0   = blockIdx.x * 16;

    __shared__ float                  adjb[16 * 32];          // shared by all heads
    __shared__ __align__(16) _Float16 Vt[8][32 * HDq];        // per-wave V tile 32x64
    __shared__ __align__(16) _Float16 Pt[8][16 * 32];         // per-wave P tile 16x32

    // Q fragments (reused for all key tiles), two K-steps of 32 over hd=64
    v16h qa[2];
    {
        const _Float16* qbase = q16 + ((size_t)(b * Nq + q0 + l16) * Dq + h * HDq);
        #pragma unroll
        for (int s = 0; s < 2; ++s) {
            v8h lo = *(const v8h*)(qbase + s * 32 + hi * 16);
            v8h up = *(const v8h*)(qbase + s * 32 + hi * 16 + 8);
            #pragma unroll
            for (int e = 0; e < 8; ++e) { qa[s][e] = lo[e]; qa[s][e + 8] = up[e]; }
        }
    }

    float m_r[8], l_r[8];
    v8f o[4];
    #pragma unroll
    for (int i = 0; i < 8; ++i) { m_r[i] = -1.0e30f; l_r[i] = 0.0f; }
    #pragma unroll
    for (int t = 0; t < 4; ++t) o[t] = (v8f){};

    const size_t adj_base = (size_t)b * Nq * Nq;

    for (int j = 0; j < 32; ++j) {           // 32 tiles x 32 keys = 1024 keys
        __syncthreads();   // previous-iteration adjb readers done
        #pragma unroll
        for (int u = 0; u < 2; ++u) {        // 512 adj elems, 2 per thread
            int idx = tid + u * 256;
            int r = idx >> 5, c = idx & 31;
            int av = adj[adj_base + (size_t)(q0 + r) * Nq + j * 32 + c];
            adjb[idx] = (av != 0) ? 0.0f : -10000.0f;
        }
        if (j < 31)        // prefetch next adj tile row for this thread
            __builtin_prefetch(&adj[adj_base + (size_t)(q0 + (tid >> 4)) * Nq + (j + 1) * 32], 0, 0);
        __syncthreads();

        // scores: S = Q * K^T for 32 keys (2 column tiles x 2 K-steps)
        v8f sc[2];
        sc[0] = (v8f){}; sc[1] = (v8f){};
        #pragma unroll
        for (int cc = 0; cc < 2; ++cc) {
            #pragma unroll
            for (int s = 0; s < 2; ++s) {
                v16h kb = *(const v16h*)(k16 + ((size_t)(b * Nq + j * 32 + cc * 16 + l16) * Dq
                                                + h * HDq + s * 32 + hi * 16));
                sc[cc] = __builtin_amdgcn_wmma_f32_16x16x32_f16(false, qa[s], false, kb,
                                                                (short)0, sc[cc], false, false);
            }
        }

        // stage V tile (32 x 64 f16) into per-wave LDS
        {
            const char* vsrc = (const char*)(v16 + ((size_t)(b * Nq + j * 32) * Dq + h * HDq));
#if USE_ASYNC_LDS
            for (int c = lane; c < 256; c += 32) {
                int row = c >> 3, ch = c & 7;
                __builtin_amdgcn_global_load_async_to_lds_b128(
                    GPTR(vsrc + (size_t)row * Dq * 2 + ch * 16),
                    LPTR((char*)Vt[wave] + 16 * c), 0, 0);
            }
            __builtin_amdgcn_s_wait_asynccnt(0);
#else
            for (int c = lane; c < 256; c += 32) {
                int row = c >> 3, ch = c & 7;
                ((uint4*)Vt[wave])[c] =
                    *(const uint4*)(vsrc + (size_t)row * Dq * 2 + ch * 16);
            }
#endif
        }

        // online softmax update (row-wise across 16 lanes of each half-wave)
        #pragma unroll
        for (int i = 0; i < 8; ++i) {
            int rr = (i + hi * 8) * 32;
            float s0 = sc[0][i] * 0.125f + adjb[rr + l16];
            float s1 = sc[1][i] * 0.125f + adjb[rr + 16 + l16];
            float tm = fmaxf(s0, s1);
            tm = fmaxf(tm, __shfl_xor(tm, 1, 16));
            tm = fmaxf(tm, __shfl_xor(tm, 2, 16));
            tm = fmaxf(tm, __shfl_xor(tm, 4, 16));
            tm = fmaxf(tm, __shfl_xor(tm, 8, 16));
            float mn   = fmaxf(m_r[i], tm);
            float corr = __expf(m_r[i] - mn);
            float p0   = __expf(s0 - mn);
            float p1   = __expf(s1 - mn);
            float ps = p0 + p1;
            ps += __shfl_xor(ps, 1, 16);
            ps += __shfl_xor(ps, 2, 16);
            ps += __shfl_xor(ps, 4, 16);
            ps += __shfl_xor(ps, 8, 16);
            l_r[i] = l_r[i] * corr + ps;
            m_r[i] = mn;
            #pragma unroll
            for (int t = 0; t < 4; ++t) o[t][i] *= corr;
            Pt[wave][rr + l16]      = (_Float16)p0;   // C-layout -> LDS
            Pt[wave][rr + 16 + l16] = (_Float16)p1;
        }

        // restripe P from C-layout to full 16x32 A-layout fragment
        v16h pa;
        {
            const _Float16* pp = &Pt[wave][l16 * 32 + hi * 16];
            v8h lo = *(const v8h*)pp;          // K 0..7  (hi lanes: 16..23)
            v8h up = *(const v8h*)(pp + 8);    // K 8..15 (hi lanes: 24..31)
            #pragma unroll
            for (int e = 0; e < 8; ++e) { pa[e] = lo[e]; pa[e + 8] = up[e]; }
        }

        // O += P (16x32) @ V (32x64), fully dense
        #pragma unroll
        for (int t = 0; t < 4; ++t) {
            v16h vb;
            #pragma unroll
            for (int e = 0; e < 16; ++e)       // B: lo lanes K=0..15, hi lanes K=16..31
                vb[e] = Vt[wave][(hi * 16 + e) * HDq + t * 16 + l16];
            o[t] = __builtin_amdgcn_wmma_f32_16x16x32_f16(false, pa, false, vb,
                                                          (short)0, o[t], false, false);
        }
    }

    // normalize and store (heads interleaved back into [B*N, D] f16)
    #pragma unroll
    for (int t = 0; t < 4; ++t) {
        #pragma unroll
        for (int i = 0; i < 8; ++i) {
            int row = b * Nq + q0 + i + hi * 8;
            int col = h * HDq + t * 16 + l16;
            attn16[(size_t)row * Dq + col] = (_Float16)(o[t][i] / l_r[i]);
        }
    }
}

// --------------------- output projection + bias + residual ------------------
// grid: (M/16) ; block 256 = 8 waves ; wave owns 4 column tiles (64 cols).
__global__ __launch_bounds__(256) void proj_kernel(
    const _Float16* __restrict__ attn16, const _Float16* __restrict__ wo16,
    const float* __restrict__ bo, const float* __restrict__ xin,
    float* __restrict__ y) {

    const int tid  = threadIdx.x;
    const int wave = tid >> 5;
    const int lane = tid & 31;
    const int hi   = lane >> 4;
    const int l16  = lane & 15;
    const int m0   = blockIdx.x * 16;

    __shared__ __align__(16) _Float16 As[16 * Dq];
    {
        const char* src = (const char*)(attn16 + (size_t)m0 * Dq);
#if USE_ASYNC_LDS
        for (int c = tid; c < 16 * Dq / 8; c += 256)
            __builtin_amdgcn_global_load_async_to_lds_b128(
                GPTR(src + 16 * c), LPTR((char*)As + 16 * c), 0, 0);
        __builtin_amdgcn_s_wait_asynccnt(0);
#else
        for (int c = tid; c < 16 * Dq / 8; c += 256)
            ((uint4*)As)[c] = ((const uint4*)src)[c];
#endif
    }
    __syncthreads();

    v8f acc[4];
    #pragma unroll
    for (int t = 0; t < 4; ++t) acc[t] = (v8f){};

    for (int kk = 0; kk < Dq; kk += 32) {
        const _Float16* ap = As + l16 * Dq + kk + hi * 16;
        v8h alo = *(const v8h*)ap;
        v8h ahi = *(const v8h*)(ap + 8);
        v16h a;
        #pragma unroll
        for (int e = 0; e < 8; ++e) { a[e] = alo[e]; a[e + 8] = ahi[e]; }
        #pragma unroll
        for (int t = 0; t < 4; ++t) {
            int ncol = (wave * 4 + t) * 16;
            v16h b = *(const v16h*)(wo16 + (size_t)(ncol + l16) * Dq + kk + hi * 16);
            acc[t] = __builtin_amdgcn_wmma_f32_16x16x32_f16(false, a, false, b,
                                                            (short)0, acc[t], false, false);
        }
    }

    #pragma unroll
    for (int t = 0; t < 4; ++t) {
        int col = (wave * 4 + t) * 16 + l16;
        float bias = bo[col];
        #pragma unroll
        for (int i = 0; i < 8; ++i) {
            size_t idx = (size_t)(m0 + i + hi * 8) * Dq + col;
            y[idx] = acc[t][i] + bias + xin[idx];   // bias + residual
        }
    }
}

// ------------------------------ LayerNorm ----------------------------------
// grid: B*N blocks ; block 256 ; one row of 512, in-place on d_out.
__global__ __launch_bounds__(256) void ln_kernel(float* __restrict__ y,
                                                 const float* __restrict__ gamma,
                                                 const float* __restrict__ beta) {
    const int tid  = threadIdx.x;
    const int wave = tid >> 5;
    const int lane = tid & 31;
    const size_t base = (size_t)blockIdx.x * Dq;

    float v0 = y[base + tid];
    float v1 = y[base + tid + 256];
    float s1 = v0 + v1;
    float s2 = v0 * v0 + v1 * v1;
    #pragma unroll
    for (int off = 16; off > 0; off >>= 1) {
        s1 += __shfl_xor(s1, off);
        s2 += __shfl_xor(s2, off);
    }

    __shared__ float red1[8], red2[8];
    if (lane == 0) { red1[wave] = s1; red2[wave] = s2; }
    __syncthreads();
    if (tid == 0) {
        float a = 0.f, q = 0.f;
        #pragma unroll
        for (int i = 0; i < 8; ++i) { a += red1[i]; q += red2[i]; }
        float mu  = a / (float)Dq;
        float var = q / (float)Dq - mu * mu;
        red1[0] = mu;
        red2[0] = rsqrtf(var + 1e-5f);
    }
    __syncthreads();
    float mu = red1[0], rstd = red2[0];
    y[base + tid]       = (v0 - mu) * rstd * gamma[tid]       + beta[tid];
    y[base + tid + 256] = (v1 - mu) * rstd * gamma[tid + 256] + beta[tid + 256];
}

// ------------------------------ launch -------------------------------------
extern "C" void kernel_launch(void* const* d_in, const int* in_sizes, int n_in,
                              void* d_out, int out_size, void* d_ws, size_t ws_size,
                              hipStream_t stream) {
    (void)in_sizes; (void)n_in; (void)out_size; (void)ws_size;

    const float* x    = (const float*)d_in[0];
    const int*   adj  = (const int*)d_in[1];
    const float* wq   = (const float*)d_in[2];
    const float* wk   = (const float*)d_in[3];
    const float* wv   = (const float*)d_in[4];
    const float* wo   = (const float*)d_in[5];
    const float* bo   = (const float*)d_in[6];
    const float* gam  = (const float*)d_in[7];
    const float* bet  = (const float*)d_in[8];
    float* out = (float*)d_out;

    // workspace layout, ~42MB total
    const size_t XN = (size_t)Bq * Nq * Dq;          // 4,194,304 elements
    const size_t WN = (size_t)Dq * Dq;               // 262,144 elements
    char* ws = (char*)d_ws;
    _Float16* x16  = (_Float16*)(ws);
    _Float16* wq16 = (_Float16*)(ws + 2 * XN);
    _Float16* wk16 = wq16 + WN;
    _Float16* wv16 = wk16 + WN;
    _Float16* wo16 = wv16 + WN;
    _Float16* q16  = wo16 + WN;
    _Float16* k16  = q16 + XN;
    _Float16* v16  = k16 + XN;
    _Float16* a16  = v16 + XN;

    // 1) f32 -> f16 conversions
    cvt_f32_f16<<<(int)(XN / 256), 256, 0, stream>>>(x, x16, (int)XN);
    cvt_f32_f16<<<(int)(WN / 256), 256, 0, stream>>>(wq, wq16, (int)WN);
    cvt_f32_f16<<<(int)(WN / 256), 256, 0, stream>>>(wk, wk16, (int)WN);
    cvt_f32_f16<<<(int)(WN / 256), 256, 0, stream>>>(wv, wv16, (int)WN);
    cvt_f32_f16<<<(int)(WN / 256), 256, 0, stream>>>(wo, wo16, (int)WN);

    // 2) Q/K/V projections: grid (8192/16, 512/128, 3)
    qkv_gemm<<<dim3(Bq * Nq / 16, Dq / 128, 3), 256, 0, stream>>>(
        x16, wq16, wk16, wv16, q16, k16, v16);

    // 3) masked flash attention: grid (64 q-tiles, 8 batches), wave = head
    attn_kernel<<<dim3(Nq / 16, Bq), 256, 0, stream>>>(q16, k16, v16, adj, a16);

    // 4) out-projection + bias + residual -> d_out (f32)
    proj_kernel<<<dim3(Bq * Nq / 16), 256, 0, stream>>>(a16, wo16, bo, x, out);

    // 5) LayerNorm in place on d_out
    ln_kernel<<<dim3(Bq * Nq), 256, 0, stream>>>(out, gam, bet);
}
